// CausalHeteroGNN_44375602102697
// MI455X (gfx1250) — compile-verified, matrix operand order checked by reading, not run on hardware
//
#include <hip/hip_runtime.h>
#include <cmath>

// ===========================================================================
// CausalHeteroGNN forward for MI455X (gfx1250, wave32, WMMA bf16->f32).
//
// Input layout (d_in):
//   0 x_drug[2048,256] 1 x_disease 2 x_gene (f32)
//   3..8 e_targets, e_rev_targets, e_treats, e_rev_treats, e_assoc, e_rev_assoc
//        (int32 [2,E], row0=src, row1=dst)
//   9 adj_dgd[2048,2048] 10 adj_dd 11 adj_dgr (f32)
//   12.. params, JAX pytree flatten (dict keys sorted ASCII):
//     +0 ca_Wo +1 ca_Wqkv +2 ca_bo +3 ca_bqkv
//     +4 enc_disease_W +5 enc_disease_b +6 enc_drug_W +7 enc_drug_b
//     +8 enc_gene_W +9 enc_gene_b
//     +10..+99 layers: L in 0..2, rel in [assoc,rev_assoc,rev_targets,
//        rev_treats,targets,treats], each [Wd,Ws,ad,as,b]  (3*6*5=90)
//     +100 ln_b +101 ln_g
//     +102 mp_Wo +103 mp_Wqkv +104 mp_bo +105 mp_bqkv
//     +106 mp_dd_W +107 mp_dd_b +108 mp_dgd_W +109 mp_dgd_b
//     +110 mp_dgr_W +111 mp_dgr_b
// Output: [drug 2048x128][disease][gene] f32.
// ===========================================================================

typedef __bf16 bf16_t;
typedef bf16_t v16bf __attribute__((ext_vector_type(16)));
typedef bf16_t v2bf  __attribute__((ext_vector_type(2)));
typedef float  v8f   __attribute__((ext_vector_type(8)));

union ABu { v16bf v; unsigned int u[8]; unsigned short s[16]; };

// Let the backend pick the native bf16 converts (v_cvt_*bf16_f32 class on
// CDNA5) instead of software shift/round sequences.
__device__ __forceinline__ unsigned short f2bf(float f) {
  union { bf16_t h; unsigned short s; } u;
  u.h = (bf16_t)f;
  return u.s;
}
__device__ __forceinline__ unsigned int pack2bf(float a, float b) {
  union { v2bf h; unsigned int u; } u;
  u.h = v2bf{(bf16_t)a, (bf16_t)b};
  return u.u;
}

__device__ __forceinline__ v8f wmma_bf16(v16bf a, v16bf b, v8f c) {
  return __builtin_amdgcn_wmma_f32_16x16x32_bf16(false, a, false, b,
                                                 (short)0, c, false, false);
}

// --------------------------------------------------------------------------
// GEMM: C[M,N] = A[M,K] @ B + bias, fp32 in HBM, bf16 WMMA with fp32 acc.
// PRECONDITIONS (all call sites satisfy): M%64==0, N%64==0, K%32==0,
// 16B-aligned rows (lda/ldb multiples of 4, bases 256B-aligned).
// transB=0: B physical [K,N]; transB=1: logical B=W^T, W physical [N,K].
// Block 128 thr = 4 waves; tile 64x64x32; LDS double-buffered, tiles
// prefetched into registers via b128 loads while WMMAs consume current tile.
// --------------------------------------------------------------------------
__global__ __launch_bounds__(128) void gemm_bf16_k(
    const float* __restrict__ A, const float* __restrict__ B,
    float* __restrict__ C, int M, int N, int K,
    int lda, int ldb, int ldc, int transB, const float* __restrict__ bias) {
  __shared__ unsigned int As32[2][64][18];   // bf16 pairs, [row][kpair]
  __shared__ unsigned int Bs32[2][64][18];   // bf16 pairs, [n][kpair]
  const int tid  = threadIdx.x;
  const int m0   = blockIdx.y * 64, n0 = blockIdx.x * 64;
  const int w    = tid >> 5, lane = tid & 31;
  const int l16  = lane & 15;
  const int hi   = (lane >> 4) & 1;
  const int lm   = (w << 4) + l16;
  v8f acc0 = {}, acc1 = {}, acc2 = {}, acc3 = {};
  float4 ra[4], rb[4];

  auto loadTiles = [&](int k0) {
#pragma unroll
    for (int q = 0; q < 4; ++q) {            // A: 64x32, 4 rows per wavefront
      int idx4 = q * 128 + tid;
      int i = idx4 >> 3, j4 = (idx4 & 7) << 2;
      ra[q] = *(const float4*)(A + (size_t)(m0 + i) * lda + k0 + j4);
    }
    if (transB) {
#pragma unroll
      for (int q = 0; q < 4; ++q) {          // W rows: coalesced along K
        int idx4 = q * 128 + tid;
        int n = idx4 >> 3, j4 = (idx4 & 7) << 2;
        rb[q] = *(const float4*)(B + (size_t)(n0 + n) * ldb + k0 + j4);
      }
    } else {
#pragma unroll
      for (int q = 0; q < 4; ++q) {          // B rows: coalesced along N
        int idx4 = q * 128 + tid;
        int n4 = (idx4 & 15) << 2, k = idx4 >> 4;
        rb[q] = *(const float4*)(B + (size_t)(k0 + k) * ldb + n0 + n4);
      }
    }
  };
  auto stashTiles = [&](int buf) {
#pragma unroll
    for (int q = 0; q < 4; ++q) {
      int idx4 = q * 128 + tid;
      int i = idx4 >> 3, j2 = (idx4 & 7) << 1;
      As32[buf][i][j2]     = pack2bf(ra[q].x, ra[q].y);
      As32[buf][i][j2 + 1] = pack2bf(ra[q].z, ra[q].w);
    }
    if (transB) {
#pragma unroll
      for (int q = 0; q < 4; ++q) {
        int idx4 = q * 128 + tid;
        int n = idx4 >> 3, j2 = (idx4 & 7) << 1;
        Bs32[buf][n][j2]     = pack2bf(rb[q].x, rb[q].y);
        Bs32[buf][n][j2 + 1] = pack2bf(rb[q].z, rb[q].w);
      }
    } else {
#pragma unroll
      for (int q = 0; q < 4; ++q) {          // register transpose into [n][k]
        int idx4 = q * 128 + tid;
        int n4 = (idx4 & 15) << 2, k = idx4 >> 4;
        ((unsigned short*)&Bs32[buf][n4 + 0][0])[k] = f2bf(rb[q].x);
        ((unsigned short*)&Bs32[buf][n4 + 1][0])[k] = f2bf(rb[q].y);
        ((unsigned short*)&Bs32[buf][n4 + 2][0])[k] = f2bf(rb[q].z);
        ((unsigned short*)&Bs32[buf][n4 + 3][0])[k] = f2bf(rb[q].w);
      }
    }
  };

  loadTiles(0);
  stashTiles(0);
  __syncthreads();
  int cur = 0;
  for (int k0 = 0; k0 < K; k0 += 32) {
    const bool nxt = (k0 + 32) < K;
    if (nxt) loadTiles(k0 + 32);             // prefetch while computing
    ABu a;
#pragma unroll
    for (int r8 = 0; r8 < 8; ++r8)           // A operand: lane<->M, regs<->K
      a.u[r8] = As32[cur][lm][((r8 & 4) ? 8 : 0) + (hi ? 4 : 0) + (r8 & 3)];
#pragma unroll
    for (int nt = 0; nt < 4; ++nt) {
      ABu b;
      const int ln = (nt << 4) + l16;
#pragma unroll
      for (int r8 = 0; r8 < 8; ++r8)         // B operand: lane<->N, regs<->K
        b.u[r8] = Bs32[cur][ln][((r8 & 4) ? 8 : 0) + (hi ? 4 : 0) + (r8 & 3)];
      if      (nt == 0) acc0 = wmma_bf16(a.v, b.v, acc0);
      else if (nt == 1) acc1 = wmma_bf16(a.v, b.v, acc1);
      else if (nt == 2) acc2 = wmma_bf16(a.v, b.v, acc2);
      else              acc3 = wmma_bf16(a.v, b.v, acc3);
    }
    if (nxt) {
      stashTiles(1 ^ cur);
      __syncthreads();
      cur ^= 1;
    }
  }
  const int row_base = m0 + (w << 4) + (hi << 3);   // C: reg r -> row r+8*hi
#pragma unroll
  for (int nt = 0; nt < 4; ++nt) {
    v8f av = (nt == 0) ? acc0 : (nt == 1) ? acc1 : (nt == 2) ? acc2 : acc3;
    const int coln = n0 + (nt << 4) + l16;
    const float bv = (bias != nullptr) ? bias[coln] : 0.f;
#pragma unroll
    for (int r = 0; r < 8; ++r)
      C[(size_t)(row_base + r) * ldc + coln] = av[r] + bv;
  }
}

// --------------------------------------------------------------------------
// Fused flash attention (seq=2048, H=4, hd=32).  qkv[2048,384] (q|k|v).
// One wave per (16-query tile, head): stream keys in chunks of 32, online
// softmax, QK^T + PV both via WMMA (hd=32 == bf16 WMMA K exactly).
// Single-wave workgroup => barrier ops degrade to S_NOP (free).
// --------------------------------------------------------------------------
__global__ __launch_bounds__(32) void flash_attn_k(
    const float* __restrict__ qkv, float* __restrict__ outp) {
  const int h = blockIdx.y;
  const int q0 = blockIdx.x * 16;
  const int lane = threadIdx.x;
  const int l16 = lane & 15;
  const int hi = (lane >> 4) & 1;
  const float sc = 0.17677669529663687f;        // 1/sqrt(32)
  __shared__ unsigned short Pl[16][36];

  ABu qa;
  {
    const float* qr = qkv + (size_t)(q0 + l16) * 384 + h * 32;
#pragma unroll
    for (int r8 = 0; r8 < 8; ++r8) {
      int base = ((r8 & 4) ? 16 : 0) + (hi ? 8 : 0) + ((r8 & 3) << 1);
      qa.u[r8] = pack2bf(qr[base], qr[base + 1]);
    }
  }
  float mrow[8], lrow[8];
#pragma unroll
  for (int r = 0; r < 8; ++r) { mrow[r] = -3.0e38f; lrow[r] = 0.f; }
  v8f o0 = {}, o1 = {};

  for (int kt = 0; kt < 2048; kt += 32) {
    ABu kb0, kb1;
    const float* kp0 = qkv + (size_t)(kt + l16) * 384 + 128 + h * 32;
    const float* kp1 = qkv + (size_t)(kt + 16 + l16) * 384 + 128 + h * 32;
#pragma unroll
    for (int r8 = 0; r8 < 8; ++r8) {
      int base = ((r8 & 4) ? 16 : 0) + (hi ? 8 : 0) + ((r8 & 3) << 1);
      kb0.u[r8] = pack2bf(kp0[base], kp0[base + 1]);
      kb1.u[r8] = pack2bf(kp1[base], kp1[base + 1]);
    }
    v8f z = {};
    v8f s0 = wmma_bf16(qa.v, kb0.v, z);
    v8f s1 = wmma_bf16(qa.v, kb1.v, z);
    float al[8];
#pragma unroll
    for (int r = 0; r < 8; ++r) {
      s0[r] *= sc; s1[r] *= sc;
      float a = fmaxf(s0[r], s1[r]);
#pragma unroll
      for (int off = 1; off < 16; off <<= 1) a = fmaxf(a, __shfl_xor(a, off, 32));
      float mn = fmaxf(mrow[r], a);
      al[r] = __expf(mrow[r] - mn);
      mrow[r] = mn;
      float p0 = __expf(s0[r] - mn), p1 = __expf(s1[r] - mn);
      Pl[r + 8 * hi][l16]      = f2bf(p0);
      Pl[r + 8 * hi][16 + l16] = f2bf(p1);
      float sr = p0 + p1;
#pragma unroll
      for (int off = 1; off < 16; off <<= 1) sr += __shfl_xor(sr, off, 32);
      lrow[r] = lrow[r] * al[r] + sr;
      o0[r] *= al[r]; o1[r] *= al[r];
    }
    __syncthreads();
    ABu pa, vb0, vb1;
    const float* vbase = qkv + (size_t)kt * 384 + 256 + h * 32;
#pragma unroll
    for (int r8 = 0; r8 < 8; ++r8) {
      int base = ((r8 & 4) ? 16 : 0) + (hi ? 8 : 0) + ((r8 & 3) << 1);
      pa.s[2 * r8]     = Pl[l16][base];
      pa.s[2 * r8 + 1] = Pl[l16][base + 1];
      vb0.u[r8] = pack2bf(vbase[(size_t)base * 384 + l16],
                          vbase[(size_t)(base + 1) * 384 + l16]);
      vb1.u[r8] = pack2bf(vbase[(size_t)base * 384 + 16 + l16],
                          vbase[(size_t)(base + 1) * 384 + 16 + l16]);
    }
    o0 = wmma_bf16(pa.v, vb0.v, o0);
    o1 = wmma_bf16(pa.v, vb1.v, o1);
    __syncthreads();
  }
#pragma unroll
  for (int r = 0; r < 8; ++r) {
    float inv = 1.f / lrow[r];
    int row = q0 + r + 8 * hi;
    outp[(size_t)row * 128 + h * 32 + l16]      = o0[r] * inv;
    outp[(size_t)row * 128 + h * 32 + 16 + l16] = o1[r] * inv;
  }
}

// ----------------------------- small kernels ------------------------------
__global__ void deg_count_k(const int* __restrict__ dst, int E, int* __restrict__ deg) {
  int i = blockIdx.x * blockDim.x + threadIdx.x;
  if (i < E) atomicAdd(&deg[dst[i]], 1);
}

__global__ __launch_bounds__(256) void scan2048_k(const int* __restrict__ deg,
                                                  int* __restrict__ rowptr) {
  __shared__ int part[256];
  int tid = threadIdx.x;
  int loc[8]; int sum = 0; int base = tid * 8;
#pragma unroll
  for (int i = 0; i < 8; ++i) { loc[i] = deg[base + i]; sum += loc[i]; }
  part[tid] = sum; __syncthreads();
  if (tid == 0) {
    int r = 0;
    for (int i = 0; i < 256; ++i) { int t = part[i]; part[i] = r; r += t; }
    rowptr[2048] = r;
  }
  __syncthreads();
  int off = part[tid];
#pragma unroll
  for (int i = 0; i < 8; ++i) { rowptr[base + i] = off; off += loc[i]; }
}

__global__ void csr_fill_k(const int* __restrict__ src, const int* __restrict__ dst,
                           int E, const int* __restrict__ rowptr,
                           int* __restrict__ cur, int* __restrict__ col) {
  int i = blockIdx.x * blockDim.x + threadIdx.x;
  if (i >= E) return;
  int d = dst[i];
  int pos = atomicAdd(&cur[d], 1);
  col[rowptr[d] + pos] = src[i];
}

// a_s[n,h] = sum_c hs[n,h*32+c]*as[h,c]; same for a_d. One block per node,
// one wave32 per head.
__global__ __launch_bounds__(128) void gat_scores_k(
    const float* __restrict__ hs, const float* __restrict__ hd,
    const float* __restrict__ aw_s, const float* __restrict__ aw_d,
    float* __restrict__ asb, float* __restrict__ adb) {
  int n = blockIdx.x, tid = threadIdx.x;
  int h = tid >> 5, lc = tid & 31;
  float ps = hs[(size_t)n * 128 + tid] * aw_s[h * 32 + lc];
  float pd = hd[(size_t)n * 128 + tid] * aw_d[h * 32 + lc];
#pragma unroll
  for (int off = 1; off < 32; off <<= 1) {
    ps += __shfl_xor(ps, off, 32);
    pd += __shfl_xor(pd, off, 32);
  }
  if (lc == 0) { asb[n * 4 + h] = ps; adb[n * 4 + h] = pd; }
}

// Per-dst GAT gather: softmax over incoming edges (+ self loop), weighted sum
// of hs rows, + relation bias.  One block per dst node, thread = channel.
// Neighbor hs rows are streamed into LDS with GLOBAL_LOAD_ASYNC_TO_LDS_B32
// (ASYNCcnt) *during* the alpha/max scan, drained with s_wait_asynccnt
// before the weighted pass.  Generic LDS pointers: low 32 bits == LDS offset.
#define GSTAGE 64
__global__ __launch_bounds__(128) void gat_gather_k(
    const float* __restrict__ hs, const float* __restrict__ asb,
    const float* __restrict__ adb, const int* __restrict__ rowptr,
    const int* __restrict__ col, const float* __restrict__ bias,
    float* __restrict__ agg, int accumulate) {
  __shared__ float rows[GSTAGE][128];
  const int d = blockIdx.x, c = threadIdx.x, h = c >> 5;
  const float adv = adb[d * 4 + h];
  const int beg = rowptr[d], end = rowptr[d + 1];
  const int deg = end - beg;
  const int nst = (deg < (GSTAGE - 1)) ? deg : (GSTAGE - 1);

  {  // async-stage self row into slot 0
    unsigned lo = (unsigned)(size_t)(&rows[0][c]);
    unsigned long long gp = (unsigned long long)(size_t)(hs + (size_t)d * 128 + c);
    asm volatile("global_load_async_to_lds_b32 %0, %1, off"
                 :: "v"(lo), "v"(gp) : "memory");
  }
  float as_self = asb[d * 4 + h] + adv;
  as_self = (as_self > 0.f) ? as_self : 0.2f * as_self;
  float mx = as_self;
  for (int j = 0; j < deg; ++j) {
    int s = col[beg + j];
    if (j < nst) {  // kick off row DMA, overlapped with the max scan
      unsigned lo = (unsigned)(size_t)(&rows[j + 1][c]);
      unsigned long long gp = (unsigned long long)(size_t)(hs + (size_t)s * 128 + c);
      asm volatile("global_load_async_to_lds_b32 %0, %1, off"
                   :: "v"(lo), "v"(gp) : "memory");
    }
    float a = asb[s * 4 + h] + adv;
    a = (a > 0.f) ? a : 0.2f * a;
    mx = fmaxf(mx, a);
  }
  asm volatile("s_wait_asynccnt 0" ::: "memory");
  float den, acc;
  {
    float w0 = __expf(as_self - mx);
    den = w0; acc = rows[0][c] * w0;
  }
  for (int j = 0; j < deg; ++j) {
    int s = col[beg + j];
    float a = asb[s * 4 + h] + adv;
    a = (a > 0.f) ? a : 0.2f * a;
    float w = __expf(a - mx);
    den += w;
    float hv = (j < nst) ? rows[j + 1][c] : hs[(size_t)s * 128 + c];
    acc += hv * w;
  }
  float o = acc / (den + 1e-16f) + bias[c];
  size_t idx = (size_t)d * 128 + c;
  if (accumulate) agg[idx] += o; else agg[idx] = o;
}

__global__ void elu_half_k(const float* __restrict__ a, float* __restrict__ o, int n) {
  int i = blockIdx.x * blockDim.x + threadIdx.x;
  if (i < n) { float z = 0.5f * a[i]; o[i] = (z > 0.f) ? z : expm1f(z); }
}

__global__ __launch_bounds__(128) void resid_ln_k(
    const float* __restrict__ x, const float* __restrict__ pj,
    const float* __restrict__ g, const float* __restrict__ bb,
    float* __restrict__ outp) {
  int n = blockIdx.x, tid = threadIdx.x;
  __shared__ float red[128];
  float v = x[(size_t)n * 128 + tid] + pj[(size_t)n * 128 + tid];
  red[tid] = v; __syncthreads();
  for (int s = 64; s > 0; s >>= 1) { if (tid < s) red[tid] += red[tid + s]; __syncthreads(); }
  float mu = red[0] * (1.f / 128.f); __syncthreads();
  float d = v - mu;
  red[tid] = d * d; __syncthreads();
  for (int s = 64; s > 0; s >>= 1) { if (tid < s) red[tid] += red[tid + s]; __syncthreads(); }
  float var = red[0] * (1.f / 128.f);
  outp[(size_t)n * 128 + tid] = d * rsqrtf(var + 1e-5f) * g[tid] + bb[tid];
}

// Metapath MHA: seq=3 per drug node, H=4 heads of 32.  One wave per head;
// 3x3 scores via shfl dot products; returns mean over the 3 query positions.
__global__ __launch_bounds__(128) void mp_attn_k(
    const float* __restrict__ qkv, float* __restrict__ outmean) {
  int n = blockIdx.x, tid = threadIdx.x;
  int h = tid >> 5, lc = tid & 31;
  const float sc = 0.17677669529663687f;
  float q[3], k[3], v[3];
#pragma unroll
  for (int i = 0; i < 3; ++i) {
    const float* row = qkv + (size_t)(n * 3 + i) * 384;
    q[i] = row[h * 32 + lc];
    k[i] = row[128 + h * 32 + lc];
    v[i] = row[256 + h * 32 + lc];
  }
  float s[3][3];
#pragma unroll
  for (int i = 0; i < 3; ++i)
#pragma unroll
    for (int j = 0; j < 3; ++j) {
      float p = q[i] * k[j];
#pragma unroll
      for (int off = 1; off < 32; off <<= 1) p += __shfl_xor(p, off, 32);
      s[i][j] = p * sc;
    }
  float om = 0.f;
#pragma unroll
  for (int i = 0; i < 3; ++i) {
    float mx = fmaxf(s[i][0], fmaxf(s[i][1], s[i][2]));
    float e0 = __expf(s[i][0] - mx), e1 = __expf(s[i][1] - mx), e2 = __expf(s[i][2] - mx);
    float den = e0 + e1 + e2;
    om += (e0 * v[0] + e1 * v[1] + e2 * v[2]) / den;
  }
  outmean[(size_t)n * 128 + tid] = om * (1.f / 3.f);
}

// ------------------------------- host side --------------------------------
static inline void gemm(hipStream_t s, const float* A, const float* B, float* C,
                        int M, int N, int K, int lda, int ldb, int ldc,
                        int transB, const float* bias) {
  dim3 g((N + 63) / 64, (M + 63) / 64);
  gemm_bf16_k<<<g, dim3(128), 0, s>>>(A, B, C, M, N, K, lda, ldb, ldc, transB, bias);
}

extern "C" void kernel_launch(void* const* d_in, const int* in_sizes, int n_in,
                              void* d_out, int out_size, void* d_ws, size_t ws_size,
                              hipStream_t stream) {
  (void)out_size; (void)ws_size;
  if (n_in < 124) return;   // unexpected flatten; nothing safe to do
  const int Nn = 2048;
  const int E = in_sizes[3] / 2;

  auto fin = [&](int i) { return (const float*)d_in[i]; };
  auto iin = [&](int i) { return (const int*)d_in[i]; };
  const int P0 = 12;
  const float* ca_Wo   = fin(P0 + 0);
  const float* ca_Wqkv = fin(P0 + 1);
  const float* ca_bo   = fin(P0 + 2);
  const float* ca_bqkv = fin(P0 + 3);
  const float* enc_W[3] = { fin(P0 + 6), fin(P0 + 4), fin(P0 + 8) }; // drug,disease,gene
  const float* enc_b[3] = { fin(P0 + 7), fin(P0 + 5), fin(P0 + 9) };
  auto lpar = [&](int L, int sortIdx, int t) { return fin(P0 + 10 + L * 30 + sortIdx * 5 + t); };
  const float* ln_b = fin(P0 + 100);
  const float* ln_g = fin(P0 + 101);
  const float* mp_Wo   = fin(P0 + 102);
  const float* mp_Wqkv = fin(P0 + 103);
  const float* mp_bo   = fin(P0 + 104);
  const float* mp_bqkv = fin(P0 + 105);
  const float* mp_W[3] = { fin(P0 + 108), fin(P0 + 106), fin(P0 + 110) }; // dgd,dd,dgr
  const float* mp_b[3] = { fin(P0 + 109), fin(P0 + 107), fin(P0 + 111) };

  // ---- workspace carve ----
  char* wsb = (char*)d_ws; size_t off = 0;
  auto alloc = [&](size_t bytes) {
    off = (off + 255) & ~(size_t)255;
    void* p = wsb + off; off += bytes; return p;
  };
  const size_t NM = (size_t)Nn * 128;
  float* xA[3], *xB[3], *agg[3];
  for (int t = 0; t < 3; ++t) xA[t] = (float*)alloc(NM * 4);
  for (int t = 0; t < 3; ++t) xB[t] = (float*)alloc(NM * 4);
  for (int t = 0; t < 3; ++t) agg[t] = (float*)alloc(NM * 4);
  float* hsB  = (float*)alloc(NM * 4);
  float* hdB  = (float*)alloc(NM * 4);
  float* asb  = (float*)alloc((size_t)Nn * 4 * 4);
  float* adb  = (float*)alloc((size_t)Nn * 4 * 4);
  float* qkvB = (float*)alloc((size_t)Nn * 384 * 4);
  float* attO = (float*)alloc(NM * 4);
  float* projB = (float*)alloc(NM * 4);
  float* mp_tmp  = (float*)alloc(NM * 4);
  float* mp_stk  = (float*)alloc((size_t)Nn * 384 * 4);
  float* mp_qkv  = (float*)alloc((size_t)Nn * 3 * 384 * 4);
  float* mp_mean = (float*)alloc(NM * 4);
  int* cursors = (int*)alloc((size_t)6 * Nn * 4);       // deg + fill cursors
  int* rowptr[6]; int* colA[6];
  for (int r = 0; r < 6; ++r) {
    rowptr[r] = (int*)alloc((size_t)(Nn + 1) * 4);
    colA[r]   = (int*)alloc((size_t)E * 4);
  }

  // relation table: {srcType, dstType, edgeInput, sortedIdxInLayerDict}
  // types: 0=drug 1=disease 2=gene
  struct Rel { int srcT, dstT, eIdx, sIdx; };
  const Rel rels[6] = {
    {0, 2, 3, 4},  // targets     drug->gene
    {2, 0, 4, 2},  // rev_targets gene->drug
    {0, 1, 5, 5},  // treats      drug->disease
    {1, 0, 6, 3},  // rev_treats  disease->drug
    {2, 1, 7, 0},  // assoc       gene->disease
    {1, 2, 8, 1},  // rev_assoc   disease->gene
  };

  // ---- build CSRs (edges are layer-invariant) ----
  hipMemsetAsync(cursors, 0, (size_t)6 * Nn * 4, stream);
  const int EB = (E + 255) / 256;
  for (int r = 0; r < 6; ++r) {
    const int* dst = iin(rels[r].eIdx) + E;
    deg_count_k<<<EB, 256, 0, stream>>>(dst, E, cursors + r * Nn);
  }
  for (int r = 0; r < 6; ++r)
    scan2048_k<<<1, 256, 0, stream>>>(cursors + r * Nn, rowptr[r]);
  hipMemsetAsync(cursors, 0, (size_t)6 * Nn * 4, stream);
  for (int r = 0; r < 6; ++r) {
    const int* src = iin(rels[r].eIdx);
    const int* dst = src + E;
    csr_fill_k<<<EB, 256, 0, stream>>>(src, dst, E, rowptr[r], cursors + r * Nn, colA[r]);
  }

  // ---- encoders: x = X @ W + b ----
  for (int t = 0; t < 3; ++t)
    gemm(stream, fin(t), enc_W[t], xA[t], Nn, 128, 256, 256, 128, 128, 0, enc_b[t]);

  float* xc[3] = { xA[0], xA[1], xA[2] };
  float* xn[3] = { xB[0], xB[1], xB[2] };

  // ---- 3 hetero GAT layers ----
  for (int L = 0; L < 3; ++L) {
    bool first[3] = { true, true, true };
    for (int r = 0; r < 6; ++r) {
      const Rel& R = rels[r];
      const float* Wd = lpar(L, R.sIdx, 0);
      const float* Ws = lpar(L, R.sIdx, 1);
      const float* ad = lpar(L, R.sIdx, 2);
      const float* as = lpar(L, R.sIdx, 3);
      const float* bb = lpar(L, R.sIdx, 4);
      gemm(stream, xc[R.srcT], Ws, hsB, Nn, 128, 128, 128, 128, 128, 0, nullptr);
      gemm(stream, xc[R.dstT], Wd, hdB, Nn, 128, 128, 128, 128, 128, 0, nullptr);
      gat_scores_k<<<Nn, 128, 0, stream>>>(hsB, hdB, as, ad, asb, adb);
      gat_gather_k<<<Nn, 128, 0, stream>>>(hsB, asb, adb, rowptr[r], colA[r], bb,
                                           agg[R.dstT], first[R.dstT] ? 0 : 1);
      first[R.dstT] = false;
    }
    for (int t = 0; t < 3; ++t)
      elu_half_k<<<(int)(NM / 256), 256, 0, stream>>>(agg[t], xn[t], (int)NM);
    for (int t = 0; t < 3; ++t) { float* tmp = xc[t]; xc[t] = xn[t]; xn[t] = tmp; }
  }

  // ---- causal self-attention per node type + residual LayerNorm ----
  float* xdrugF = agg[0];   // reuse
  for (int t = 0; t < 3; ++t) {
    gemm(stream, xc[t], ca_Wqkv, qkvB, Nn, 384, 128, 128, 128, 384, 1, ca_bqkv);
    flash_attn_k<<<dim3(Nn / 16, 4), 32, 0, stream>>>(qkvB, attO);
    gemm(stream, attO, ca_Wo, projB, Nn, 128, 128, 128, 128, 128, 1, ca_bo);
    float* dest = (t == 0) ? xdrugF : (float*)d_out + (size_t)t * NM;
    resid_ln_k<<<Nn, 128, 0, stream>>>(xc[t], projB, ln_g, ln_b, dest);
  }

  // ---- metapath aggregator (drug) ----
  for (int i = 0; i < 3; ++i) {
    gemm(stream, xdrugF, mp_W[i], mp_tmp, Nn, 128, 128, 128, 128, 128, 0, mp_b[i]);
    gemm(stream, fin(9 + i), mp_tmp, mp_stk + i * 128, Nn, 128, Nn, Nn, 128, 384, 0, nullptr);
  }
  gemm(stream, mp_stk, mp_Wqkv, mp_qkv, Nn * 3, 384, 128, 128, 128, 384, 1, mp_bqkv);
  mp_attn_k<<<Nn, 128, 0, stream>>>(mp_qkv, mp_mean);
  gemm(stream, mp_mean, mp_Wo, (float*)d_out, Nn, 128, 128, 128, 128, 128, 1, mp_bo);
}
